// LlamaAttention_22668837388955
// MI455X (gfx1250) — matrix operand order repaired and checked by reading, not compiled, not run on hardware
//
#include <hip/hip_runtime.h>

// ---------------------------------------------------------------------------
// LlamaAttention for MI455X (gfx1250): bf16 WMMA GEMMs + flash attention.
// B=2, S=2048, D=2048, H=16, hd=128.  All matmuls via v_wmma_f32_16x16x32_bf16.
// Data staging uses GLOBAL_LOAD_ASYNC_TO_LDS (ASYNCcnt) when available.
// ---------------------------------------------------------------------------

typedef unsigned short u16t;
typedef __attribute__((ext_vector_type(16))) __bf16          v16bf;
typedef __attribute__((ext_vector_type(16))) unsigned short  v16u;
typedef __attribute__((ext_vector_type(8)))  float           v8f;
typedef __attribute__((ext_vector_type(4)))  int             v4i;

typedef __attribute__((address_space(1))) v4i* gptr_v4i;   // global (AS1)
typedef __attribute__((address_space(3))) v4i* lptr_v4i;   // LDS (AS3)

#define BATCH 2
#define SEQ   2048
#define DMODEL 2048
#define HEADS 16
#define HD    128
#define MTOT  (BATCH * SEQ)   // 4096

#if __has_builtin(__builtin_amdgcn_global_load_async_to_lds_b128)
#define HAVE_ASYNC_LDS 1
#else
#define HAVE_ASYNC_LDS 0
#endif

// Per-lane 16B copy global -> LDS. Async path bypasses VGPRs (ASYNCcnt).
__device__ __forceinline__ void cp16_g2l(const u16t* __restrict__ g, u16t* l) {
#if HAVE_ASYNC_LDS
    __builtin_amdgcn_global_load_async_to_lds_b128(
        (gptr_v4i)(u16t*)g, (lptr_v4i)l, 0, 0);
#else
    *(uint4*)l = *(const uint4*)g;
#endif
}

__device__ __forceinline__ void wait_async_lds() {
#if HAVE_ASYNC_LDS
#if __has_builtin(__builtin_amdgcn_s_wait_asynccnt)
    __builtin_amdgcn_s_wait_asynccnt(0);
#else
    asm volatile("s_wait_asynccnt 0x0" ::: "memory");
#endif
#endif
}

// f32 -> bf16 bits, round-to-nearest-even
__device__ __forceinline__ u16t f2bf(float f) {
    unsigned u = __float_as_uint(f);
    u += 0x7FFFu + ((u >> 16) & 1u);
    return (u16t)(u >> 16);
}

__device__ __forceinline__ v16bf bits2bf(v16u v) {
    return __builtin_bit_cast(v16bf, v);
}

// A-matrix (16x32, 16-bit) element->K mapping per ISA 7.12.2.
__device__ __forceinline__ int akid(int e, int half) {
    return (e < 8 ? e : e + 8) + 8 * half;
}
// B-matrix (32x16, 16-bit): lane = N, element e -> K same as A (table order)
__device__ __forceinline__ int bkid(int e, int half) {
    return (e < 8 ? e : e + 8) + 8 * half;
}

__device__ __forceinline__ v8f wmma_bf16(v16u a, v16u b, v8f c) {
    return __builtin_amdgcn_wmma_f32_16x16x32_bf16(
        false, bits2bf(a), false, bits2bf(b), (short)0, c, false, false);
}

// ---------------------------------------------------------------------------
// f32 -> bf16 conversion
// ---------------------------------------------------------------------------
__global__ void cvt_f32_bf16(const float* __restrict__ in, u16t* __restrict__ out, int n) {
    for (int i = blockIdx.x * blockDim.x + threadIdx.x; i < n; i += gridDim.x * blockDim.x)
        out[i] = f2bf(in[i]);
}

// ---------------------------------------------------------------------------
// TN GEMM:  C[m,n] = sum_k A[m,k] * W[n,k]   (M=4096, N=2048, K=2048)
// mode 0: write f32 row-major to outF
// mode 1: write bf16 to outB in [B,H,S,hd] layout (for Q/K/V)
// Block 256 thr = 8 waves (2 x 4); block tile 64x128; wave tile 32x32.
// Double-buffered LDS; async loads for tile k+1 overlap WMMAs on tile k.
// ---------------------------------------------------------------------------
__global__ __launch_bounds__(256) void gemm_tn(const u16t* __restrict__ A,
                                               const u16t* __restrict__ W,
                                               float* __restrict__ outF,
                                               u16t* __restrict__ outB,
                                               int mode) {
    __shared__ u16t sA[2][64][40];    // 80B rows keep 16B-aligned chunks
    __shared__ u16t sW[2][128][40];

    const int K = DMODEL, N = DMODEL;
    const int n0 = blockIdx.x * 128;
    const int m0 = blockIdx.y * 64;

    const int t    = threadIdx.x;
    const int wave = t >> 5, lane = t & 31;
    const int half = lane >> 4, l16 = lane & 15;
    const int waveM = wave >> 2, waveN = wave & 3;

    // staging assignments: sA 256 chunks (1/thread), sW 512 chunks (2/thread)
    const int rowA = t >> 2,  chA = (t & 3) * 8;
    const int rowW = t >> 2,  chW = (t & 3) * 8;   // second chunk: rowW + 64

    auto stage = [&](int kt, int buf) {
        const int k0 = kt * 32;
        cp16_g2l(&A[(size_t)(m0 + rowA) * K + k0 + chA], &sA[buf][rowA][chA]);
        cp16_g2l(&W[(size_t)(n0 + rowW) * K + k0 + chW], &sW[buf][rowW][chW]);
        cp16_g2l(&W[(size_t)(n0 + 64 + rowW) * K + k0 + chW], &sW[buf][64 + rowW][chW]);
    };

    v8f acc00 = {}, acc01 = {}, acc10 = {}, acc11 = {};

    stage(0, 0);
    int buf = 0;
    for (int kt = 0; kt < K / 32; ++kt) {
        wait_async_lds();
        __syncthreads();
        if (kt + 1 < K / 32) stage(kt + 1, buf ^ 1);

        v16u a0, a1, b0, b1;
#pragma unroll
        for (int e = 0; e < 16; ++e) {
            const int ka = akid(e, half);
            const int kb = bkid(e, half);
            a0[e] = sA[buf][waveM * 32 + l16][ka];
            a1[e] = sA[buf][waveM * 32 + 16 + l16][ka];
            b0[e] = sW[buf][waveN * 32 + l16][kb];
            b1[e] = sW[buf][waveN * 32 + 16 + l16][kb];
        }
        acc00 = wmma_bf16(a0, b0, acc00);
        acc01 = wmma_bf16(a0, b1, acc01);
        acc10 = wmma_bf16(a1, b0, acc10);
        acc11 = wmma_bf16(a1, b1, acc11);
        buf ^= 1;
    }

    const int gnA = n0 + waveN * 32 + l16;
    const int gnB = gnA + 16;
    if (mode == 0) {
#pragma unroll
        for (int j = 0; j < 8; ++j) {
            const int gm0 = m0 + waveM * 32 + j + 8 * half;
            const int gm1 = gm0 + 16;
            outF[(size_t)gm0 * N + gnA] = acc00[j];
            outF[(size_t)gm0 * N + gnB] = acc01[j];
            outF[(size_t)gm1 * N + gnA] = acc10[j];
            outF[(size_t)gm1 * N + gnB] = acc11[j];
        }
    } else {
        const int hA = gnA >> 7, dA = gnA & 127;
        const int hB = gnB >> 7, dB = gnB & 127;
#pragma unroll
        for (int j = 0; j < 8; ++j) {
            const int gm0 = m0 + waveM * 32 + j + 8 * half;
            const int gm1 = gm0 + 16;
            const int b0i = gm0 >> 11, s0 = gm0 & 2047;
            const int b1i = gm1 >> 11, s1 = gm1 & 2047;
            outB[(((size_t)b0i * HEADS + hA) * SEQ + s0) * HD + dA] = f2bf(acc00[j]);
            outB[(((size_t)b0i * HEADS + hB) * SEQ + s0) * HD + dB] = f2bf(acc01[j]);
            outB[(((size_t)b1i * HEADS + hA) * SEQ + s1) * HD + dA] = f2bf(acc10[j]);
            outB[(((size_t)b1i * HEADS + hB) * SEQ + s1) * HD + dB] = f2bf(acc11[j]);
        }
    }
}

// ---------------------------------------------------------------------------
// Flash attention: one block per (b,h, 128 q-rows). 8 waves, 16 q-rows/wave.
// K/V tiles (32 keys x 128) staged in LDS once per block, shared by all waves.
// Online softmax with per-row running max/sum held in the C-layout lane slots.
// ---------------------------------------------------------------------------
__global__ __launch_bounds__(256) void attn_flash(const u16t* __restrict__ qb,
                                                  const u16t* __restrict__ kb,
                                                  const u16t* __restrict__ vb,
                                                  u16t* __restrict__ ctxb) {
    __shared__ u16t sK[32][136];
    __shared__ u16t sV[32][136];
    __shared__ u16t sQ[8][16][136];
    __shared__ u16t sP[8][16][40];

    const int t    = threadIdx.x;
    const int w    = t >> 5, lane = t & 31;
    const int half = lane >> 4, l16 = lane & 15;
    const int bh   = blockIdx.y;                       // b*H + h
    const int qBase = (blockIdx.x * 8 + w) * 16;

    // ---- stage this wave's 16x128 Q tile and build 4 A fragments (hd chunks)
    for (int it = 0; it < 8; ++it) {
        const int idx = it * 32 + lane;                // 0..255 chunks
        const int row = idx >> 4, ch = (idx & 15) * 8;
        cp16_g2l(&qb[((size_t)bh * SEQ + qBase + row) * HD + ch], &sQ[w][row][ch]);
    }
    wait_async_lds();
    __syncthreads();

    v16u aQ[4];
#pragma unroll
    for (int kc = 0; kc < 4; ++kc)
#pragma unroll
        for (int e = 0; e < 16; ++e)
            aQ[kc][e] = sQ[w][l16][kc * 32 + akid(e, half)];

    float m[8], lsum[8];
    v8f acc[8];
#pragma unroll
    for (int j = 0; j < 8; ++j) { m[j] = -1e30f; lsum[j] = 0.f; }
#pragma unroll
    for (int n = 0; n < 8; ++n) acc[n] = v8f{};

    const float sc = 0.08838834764831845f;  // 1/sqrt(128)

    for (int kt = 0; kt < SEQ / 32; ++kt) {
        const int kBase = kt * 32;
        __syncthreads();
        // ---- stage 32x128 K and V tiles cooperatively (512 16B-chunks each)
        for (int it = 0; it < 2; ++it) {
            const int idx = it * 256 + t;              // 0..511
            const int row = idx >> 4, ch = (idx & 15) * 8;
            cp16_g2l(&kb[((size_t)bh * SEQ + kBase + row) * HD + ch], &sK[row][ch]);
            cp16_g2l(&vb[((size_t)bh * SEQ + kBase + row) * HD + ch], &sV[row][ch]);
            if (kt + 1 < SEQ / 32) {
                __builtin_prefetch(&kb[((size_t)bh * SEQ + kBase + 32 + row) * HD + ch], 0, 1);
                __builtin_prefetch(&vb[((size_t)bh * SEQ + kBase + 32 + row) * HD + ch], 0, 1);
            }
        }
        wait_async_lds();
        __syncthreads();

        // ---- scores: 16 q-rows x 32 keys = two 16x16 C tiles, K-loop over hd
        v8f c0 = {}, c1 = {};
#pragma unroll
        for (int kc = 0; kc < 4; ++kc) {
            v16u b0, b1;
#pragma unroll
            for (int e = 0; e < 16; ++e) {
                const int kk = kc * 32 + bkid(e, half);
                b0[e] = sK[l16][kk];
                b1[e] = sK[16 + l16][kk];
            }
            c0 = wmma_bf16(aQ[kc], b0, c0);
            c1 = wmma_bf16(aQ[kc], b1, c1);
        }

        // ---- online softmax update (row = j + 8*half; reduce over 16 lanes)
        float al[8];
#pragma unroll
        for (int j = 0; j < 8; ++j) {
            const float s0 = c0[j] * sc;
            const float s1 = c1[j] * sc;
            float r = fmaxf(s0, s1);
            r = fmaxf(r, __shfl_xor(r, 1, 32));
            r = fmaxf(r, __shfl_xor(r, 2, 32));
            r = fmaxf(r, __shfl_xor(r, 4, 32));
            r = fmaxf(r, __shfl_xor(r, 8, 32));
            const float mn = fmaxf(m[j], r);
            al[j] = __expf(m[j] - mn);
            m[j]  = mn;
            const float p0 = __expf(s0 - mn);
            const float p1 = __expf(s1 - mn);
            float rs = p0 + p1;
            rs += __shfl_xor(rs, 1, 32);
            rs += __shfl_xor(rs, 2, 32);
            rs += __shfl_xor(rs, 4, 32);
            rs += __shfl_xor(rs, 8, 32);
            lsum[j] = lsum[j] * al[j] + rs;
            sP[w][j + 8 * half][l16]      = f2bf(p0);
            sP[w][j + 8 * half][16 + l16] = f2bf(p1);
        }
#pragma unroll
        for (int n = 0; n < 8; ++n)
#pragma unroll
            for (int j = 0; j < 8; ++j) acc[n][j] *= al[j];
        __syncthreads();

        // ---- ctx += P(16x32) @ V(32x128): 8 WMMAs over hd
        v16u aP;
#pragma unroll
        for (int e = 0; e < 16; ++e)
            aP[e] = sP[w][l16][akid(e, half)];
#pragma unroll
        for (int nc = 0; nc < 8; ++nc) {
            v16u bV;
#pragma unroll
            for (int e = 0; e < 16; ++e)
                bV[e] = sV[bkid(e, half)][nc * 16 + l16];
            acc[nc] = wmma_bf16(aP, bV, acc[nc]);
        }
    }

    // ---- epilogue: normalize by lsum, store ctx bf16 in [B,S,D] layout
    const int b = bh >> 4, h = bh & 15;
    float inv[8];
#pragma unroll
    for (int j = 0; j < 8; ++j) inv[j] = 1.0f / lsum[j];
#pragma unroll
    for (int nc = 0; nc < 8; ++nc) {
        const int col = h * HD + nc * 16 + l16;
#pragma unroll
        for (int j = 0; j < 8; ++j) {
            const int gs = qBase + j + 8 * half;
            ctxb[((size_t)b * SEQ + gs) * DMODEL + col] = f2bf(acc[nc][j] * inv[j]);
        }
    }
}

// ---------------------------------------------------------------------------
// Launcher
// ---------------------------------------------------------------------------
extern "C" void kernel_launch(void* const* d_in, const int* in_sizes, int n_in,
                              void* d_out, int out_size, void* d_ws, size_t ws_size,
                              hipStream_t stream) {
    const float* x  = (const float*)d_in[0];
    const float* wq = (const float*)d_in[1];
    const float* wk = (const float*)d_in[2];
    const float* wv = (const float*)d_in[3];
    const float* wo = (const float*)d_in[4];
    float* out = (float*)d_out;

    const size_t NX = (size_t)BATCH * SEQ * DMODEL;   // 8388608
    const size_t NW = (size_t)DMODEL * DMODEL;        // 4194304

    u16t* ws  = (u16t*)d_ws;
    u16t* xb  = ws;
    u16t* wqb = xb  + NX;
    u16t* wkb = wqb + NW;
    u16t* wvb = wkb + NW;
    u16t* wob = wvb + NW;
    u16t* qb  = wob + NW;
    u16t* kb  = qb  + NX;
    u16t* vb  = kb  + NX;
    u16t* ctx = vb  + NX;

    cvt_f32_bf16<<<1024, 256, 0, stream>>>(x,  xb,  (int)NX);
    cvt_f32_bf16<<<1024, 256, 0, stream>>>(wq, wqb, (int)NW);
    cvt_f32_bf16<<<1024, 256, 0, stream>>>(wk, wkb, (int)NW);
    cvt_f32_bf16<<<1024, 256, 0, stream>>>(wv, wvb, (int)NW);
    cvt_f32_bf16<<<1024, 256, 0, stream>>>(wo, wob, (int)NW);

    dim3 ggrid(DMODEL / 128, MTOT / 64);              // 16 x 64
    gemm_tn<<<ggrid, 256, 0, stream>>>(xb, wqb, nullptr, qb, 1);
    gemm_tn<<<ggrid, 256, 0, stream>>>(xb, wkb, nullptr, kb, 1);
    gemm_tn<<<ggrid, 256, 0, stream>>>(xb, wvb, nullptr, vb, 1);

    attn_flash<<<dim3(SEQ / 128, BATCH * HEADS), 256, 0, stream>>>(qb, kb, vb, ctx);

    gemm_tn<<<ggrid, 256, 0, stream>>>(ctx, wob, out, nullptr, 0);
}
